// ourModel_56453050138989
// MI455X (gfx1250) — matrix operand order repaired and checked by reading, not compiled, not run on hardware
//
#include <hip/hip_runtime.h>
#include <hip/hip_bf16.h>

// ---------------------------------------------------------------------------
// Problem constants (match reference setup_inputs)
// ---------------------------------------------------------------------------
#define NN   50000
#define EE   800000
#define DIN  1024
#define DH   128
#define NC   40
#define EPSB 1e-5f

typedef __attribute__((ext_vector_type(16))) __bf16 v16bf;
typedef __attribute__((ext_vector_type(8)))  float  v8f;

// ---------------------------------------------------------------------------
// Small helper kernels
// ---------------------------------------------------------------------------
__global__ void zero_f32(float* __restrict__ p, int count) {
    int i = blockIdx.x * blockDim.x + threadIdx.x;
    if (i < count) p[i] = 0.0f;
}

__global__ void deg_kernel(const int* __restrict__ tgt, float* __restrict__ deg, int e) {
    int i = blockIdx.x * blockDim.x + threadIdx.x;
    if (i < e) atomicAdd(&deg[tgt[i]], 1.0f);
}

__global__ void dinv_kernel(const float* __restrict__ deg, float* __restrict__ dinv, int n) {
    int i = blockIdx.x * blockDim.x + threadIdx.x;
    if (i < n) {
        float d = deg[i];
        dinv[i] = (d > 0.0f) ? rsqrtf(d) : 0.0f;
    }
}

__global__ void norm_kernel(const int* __restrict__ src, const int* __restrict__ tgt,
                            const float* __restrict__ dinv, float* __restrict__ nrm, int e) {
    int i = blockIdx.x * blockDim.x + threadIdx.x;
    if (i < e) nrm[i] = dinv[src[i]] * dinv[tgt[i]];
}

// ---------------------------------------------------------------------------
// WMMA bf16 GEMM:  Y[Nrows,Mout] = X[Nrows,K] * W[K,Mout] + bias
//
// Block = 256 threads = 8 wave32s.  Block tile = 128 rows x 32 cols.
// Wave w owns rows [w*16, w*16+16) and BOTH 16-col tiles: one A fragment,
// two B fragments, two v_wmma per K-step (K-step = 32).
//
// LDS tiles are staged directly in WMMA *fragment order* so each lane reads
// its whole 16-element bf16 fragment as one aligned 32-byte load:
//   A (16x32 bf16, ISA 7.12.2): elem(row r, k) -> half=(k>>3)&1, j=(k&7)+8*(k>>4)
//   B (32x16 bf16):             elem(k, col)   -> half=(k>>3)&1, j=(k&7)+8*(k>>4)
// slot(tile,half,lane15) = ((tile*2+half)*16 + lane15) ; 16 bf16 per slot.
//
// Uniform per-block "fullTile" fast path: 390/391 blocks run with no per-row
// bounds checks in staging or epilogue; stores use pointer increments.
// ---------------------------------------------------------------------------
#define TILE_R 128
#define TILE_C 32
#define TILE_K 32

__device__ __forceinline__ unsigned pack2_bf16(float lo, float hi) {
    unsigned short ul = __builtin_bit_cast(unsigned short, (__bf16)lo);
    unsigned short uh = __builtin_bit_cast(unsigned short, (__bf16)hi);
    return (unsigned)ul | ((unsigned)uh << 16);
}

__global__ __launch_bounds__(256)
void gemm_bf16_wmma(const float* __restrict__ X, const float* __restrict__ W,
                    const float* __restrict__ bias, float* __restrict__ Y,
                    int Nrows, int K, int Mout)
{
    // A: 8 row-tiles * 2 halves * 16 lanes * 8 u32  = 2048 u32 (8 KB)
    // B: 2 col-tiles * 2 halves * 16 lanes * 8 u32  =  512 u32 (2 KB)
    __shared__ __align__(32) unsigned AsU[TILE_R / 16 * 2 * 16 * 8];
    __shared__ __align__(32) unsigned BsU[TILE_C / 16 * 2 * 16 * 8];

    const int tid  = threadIdx.x;
    const int lane = tid & 31;
    const int wave = tid >> 5;              // 0..7 -> row tile
    const int rowBase = blockIdx.x * TILE_R;
    const int colBase = blockIdx.y * TILE_C;
    const bool fullTile = (rowBase + TILE_R) <= Nrows;   // uniform per block

    const int half = lane >> 4;             // lane half selects K-range
    const int l15  = lane & 15;

    // ---- A staging constants: thread -> (r0 = tid>>4, kp = tid&15) ----
    const int r0  = tid >> 4;               // 0..15 ; rows r0 + 16*i
    const int kpa = tid & 15;               // 0..15 ; k = 2*kpa
    const int ka  = kpa * 2;
    const int ha  = (ka >> 3) & 1;
    const int ja  = (ka & 7) + 8 * (ka >> 4);
    // LDS u32 index for iteration i:  i*256 + (ha*16 + r0)*8 + ja/2
    const int aLds0 = (ha * 16 + r0) * 8 + (ja >> 1);
    const long long aStride = (long long)16 * K;         // rows step by 16
    const float* aBase = X + (long long)(rowBase + r0) * K + ka;

    // ---- B staging constants: thread -> (c = tid&31, kp = kp0 + 8*i) ----
    const int cB  = tid & 31;
    const int kp0 = tid >> 5;               // 0..7
    const int gcB = colBase + cB;
    const bool bValid = (gcB < Mout);
    const int gcBs = bValid ? gcB : 0;

    v8f acc0 = {};
    v8f acc1 = {};

    for (int kk = 0; kk < K; kk += TILE_K) {
        // ---------- stage A tile (128 rows x 32 k) ----------
        {
            float2 vals[8];
            if (fullTile) {
                const float* p = aBase + kk;
#pragma unroll
                for (int i = 0; i < 8; ++i) {
                    vals[i] = *(const float2*)(p);
                    p += aStride;
                }
            } else {
#pragma unroll
                for (int i = 0; i < 8; ++i) {
                    int row = rowBase + r0 + 16 * i;
                    bool valid = (row < Nrows);
                    long long off = (long long)(valid ? row : 0) * K + kk + ka;
                    float2 t = *(const float2*)(X + off);
                    vals[i].x = valid ? t.x : 0.0f;
                    vals[i].y = valid ? t.y : 0.0f;
                }
            }
#pragma unroll
            for (int i = 0; i < 8; ++i)
                AsU[aLds0 + i * 256] = pack2_bf16(vals[i].x, vals[i].y);
        }
        // ---------- stage B tile (32 k x 32 cols), branch-free ----------
        {
            float lo[2], hi[2];
#pragma unroll
            for (int i = 0; i < 2; ++i) {
                int kp = kp0 + 8 * i;
                int k  = kp * 2;
                float l = W[(long long)(kk + k) * Mout + gcBs];
                float h = W[(long long)(kk + k + 1) * Mout + gcBs];
                lo[i] = bValid ? l : 0.0f;
                hi[i] = bValid ? h : 0.0f;
            }
#pragma unroll
            for (int i = 0; i < 2; ++i) {
                int kp = kp0 + 8 * i;
                int k  = kp * 2;
                int h  = (k >> 3) & 1;
                int j  = (k & 7) + 8 * (k >> 4);
                int slot = ((cB >> 4) * 2 + h) * 16 + (cB & 15);
                BsU[slot * 8 + (j >> 1)] = pack2_bf16(lo[i], hi[i]);
            }
        }
        __syncthreads();

        // ---- one aligned 32B fragment load each, two WMMAs ----
        const v16bf* AsV = (const v16bf*)AsU;
        const v16bf* BsV = (const v16bf*)BsU;
        v16bf afrag = AsV[(wave * 2 + half) * 16 + l15];
        v16bf b0    = BsV[(0 * 2 + half) * 16 + l15];
        v16bf b1    = BsV[(1 * 2 + half) * 16 + l15];

        acc0 = __builtin_amdgcn_wmma_f32_16x16x32_bf16(
                false, afrag, false, b0, (short)0, acc0, false, false);
        acc1 = __builtin_amdgcn_wmma_f32_16x16x32_bf16(
                false, afrag, false, b1, (short)0, acc1, false, false);
        __syncthreads();
    }

    // ---- store D (16x16 f32 layout: VGPR vv -> M = vv + 8*(lane>=16), N = lane&15)
    const int row0 = rowBase + wave * 16 + 8 * half;
#pragma unroll
    for (int ct = 0; ct < 2; ++ct) {
        int col = colBase + ct * 16 + l15;
        if (col < Mout) {
            float bv = bias[col];
            float* yp = Y + (long long)row0 * Mout + col;
            const v8f& acc = (ct == 0) ? acc0 : acc1;
            if (fullTile) {
#pragma unroll
                for (int vv = 0; vv < 8; ++vv) {
                    *yp = acc[vv] + bv;
                    yp += Mout;
                }
            } else {
                int rowsLeft = Nrows - row0;
#pragma unroll
                for (int vv = 0; vv < 8; ++vv) {
                    if (vv < rowsLeft) *yp = acc[vv] + bv;
                    yp += Mout;
                }
            }
        }
    }
}

// ---------------------------------------------------------------------------
// Encoder: enc[n,k] = relu( sum_d h[n,d]*eW[lab,d,k] + eb[lab,k] ), 0 if lab<0
// one thread per (node, k)
// ---------------------------------------------------------------------------
__global__ void encoder_kernel(const float* __restrict__ h, const int* __restrict__ labels,
                               const float* __restrict__ eW, const float* __restrict__ eb,
                               float* __restrict__ enc, int n, int dout)
{
    int idx = blockIdx.x * blockDim.x + threadIdx.x;
    if (idx >= n * 4) return;
    int node = idx >> 2, k = idx & 3;
    int lab = labels[node];
    float out = 0.0f;
    if (lab >= 0) {
        const float* w  = eW + ((long long)lab * dout) * 4 + k;
        const float* hp = h + (long long)node * dout;
        float s = eb[lab * 4 + k];
        for (int d = 0; d < dout; ++d) s += hp[d] * w[(long long)d * 4];
        out = fmaxf(s, 0.0f);
    }
    enc[idx] = out;
}

// ---------------------------------------------------------------------------
// Edge message + per-class decoder + scatter-add.  One block per edge;
// lane d computes relu(db[c,d] + sum_j msg_j * dW[c,j,d]) and atomically
// accumulates into agg[tgt].  agg (25.6MB) is L2-resident on MI455X.
// ---------------------------------------------------------------------------
__global__ void edge_decoder_scatter(const float* __restrict__ enc,
                                     const float* __restrict__ nrm,
                                     const int* __restrict__ src,
                                     const int* __restrict__ tgt,
                                     const int* __restrict__ labels,
                                     const float* __restrict__ dW,
                                     const float* __restrict__ db,
                                     float* __restrict__ agg,
                                     int dout, int e)
{
    int ei = blockIdx.x;
    if (ei >= e) return;
    int s = src[ei], t = tgt[ei];
    int c = labels[t];                      // targets always labeled
    float nv = nrm[ei];
    float m0 = enc[s * 4 + 0] * nv;
    float m1 = enc[s * 4 + 1] * nv;
    float m2 = enc[s * 4 + 2] * nv;
    float m3 = enc[s * 4 + 3] * nv;
    const float* w  = dW + (long long)c * 4 * dout;
    const float* bb = db + (long long)c * dout;
    for (int d = threadIdx.x; d < dout; d += blockDim.x) {
        float y = bb[d] + m0 * w[d] + m1 * w[dout + d] + m2 * w[2 * dout + d] + m3 * w[3 * dout + d];
        y = fmaxf(y, 0.0f);
        atomicAdd(&agg[(long long)t * dout + d], y);
    }
}

// ---------------------------------------------------------------------------
// update_node: h = relu(h + agg / max(deg,1))    (in place)
// ---------------------------------------------------------------------------
__global__ void update_kernel(float* __restrict__ h, const float* __restrict__ agg,
                              const float* __restrict__ deg, int dout)
{
    int node = blockIdx.x;
    float inv = 1.0f / fmaxf(deg[node], 1.0f);
    for (int d = threadIdx.x; d < dout; d += blockDim.x) {
        long long i = (long long)node * dout + d;
        h[i] = fmaxf(h[i] + agg[i] * inv, 0.0f);
    }
}

__global__ void final_kernel(const float* __restrict__ h, const float* __restrict__ agg,
                             const float* __restrict__ deg, float* __restrict__ out, int dout)
{
    int node = blockIdx.x;
    float inv = 1.0f / fmaxf(deg[node], 1.0f);
    for (int d = threadIdx.x; d < dout; d += blockDim.x) {
        long long i = (long long)node * dout + d;
        out[i] = fmaxf(h[i] + agg[i] * inv, 0.0f);
    }
}

// ---------------------------------------------------------------------------
// BatchNorm (training stats): column sums via per-block partials + atomics
// ---------------------------------------------------------------------------
__global__ void bn_stats(const float* __restrict__ h, float* __restrict__ stats,
                         int n, int dout)
{
    int col = threadIdx.x;
    if (col >= dout) return;
    float s = 0.0f, s2 = 0.0f;
    for (int r = blockIdx.x; r < n; r += gridDim.x) {
        float v = h[(long long)r * dout + col];
        s += v; s2 += v * v;
    }
    atomicAdd(&stats[col], s);
    atomicAdd(&stats[dout + col], s2);
}

__global__ void bn_apply(float* __restrict__ h, const float* __restrict__ stats,
                         const float* __restrict__ g, const float* __restrict__ be,
                         int n, int dout)
{
    int node = blockIdx.x;
    float invN = 1.0f / (float)n;
    for (int d = threadIdx.x; d < dout; d += blockDim.x) {
        float mean = stats[d] * invN;
        float var  = stats[dout + d] * invN - mean * mean;
        float is   = rsqrtf(var + EPSB);
        long long i = (long long)node * dout + d;
        h[i] = (h[i] - mean) * is * g[d] + be[d];
    }
}

// ---------------------------------------------------------------------------
// Launch
// ---------------------------------------------------------------------------
extern "C" void kernel_launch(void* const* d_in, const int* in_sizes, int n_in,
                              void* d_out, int out_size, void* d_ws, size_t ws_size,
                              hipStream_t stream)
{
    (void)in_sizes; (void)n_in; (void)out_size; (void)ws_size;

    const float* x      = (const float*)d_in[0];
    const int*   labels = (const int*)  d_in[1];
    const int*   src    = (const int*)  d_in[2];
    const int*   tgt    = (const int*)  d_in[3];
    const float* W0  = (const float*)d_in[4];  const float* b0  = (const float*)d_in[5];
    const float* eW0 = (const float*)d_in[6];  const float* eb0 = (const float*)d_in[7];
    const float* dW0 = (const float*)d_in[8];  const float* db0 = (const float*)d_in[9];
    const float* W1  = (const float*)d_in[10]; const float* b1  = (const float*)d_in[11];
    const float* eW1 = (const float*)d_in[12]; const float* eb1 = (const float*)d_in[13];
    const float* dW1 = (const float*)d_in[14]; const float* db1 = (const float*)d_in[15];
    const float* W2  = (const float*)d_in[16]; const float* b2  = (const float*)d_in[17];
    const float* eW2 = (const float*)d_in[18]; const float* eb2 = (const float*)d_in[19];
    const float* dW2 = (const float*)d_in[20]; const float* db2 = (const float*)d_in[21];
    const float* g0  = (const float*)d_in[22]; const float* be0 = (const float*)d_in[23];
    const float* g1  = (const float*)d_in[24]; const float* be1 = (const float*)d_in[25];
    float* out = (float*)d_out;

    // workspace layout (floats)
    float* ws    = (float*)d_ws;
    float* deg   = ws;                                  // N
    float* dinv  = deg  + NN;                           // N
    float* nrm   = dinv + NN;                           // E
    float* h0    = nrm  + EE;                           // N*128
    float* h1    = h0   + (size_t)NN * DH;              // N*128
    float* enc   = h1   + (size_t)NN * DH;              // N*4
    float* stats = enc  + (size_t)NN * 4;               // 2*128

    const int B = 256;
    dim3 gemmBlk(256);
    dim3 gemmGrdH((NN + TILE_R - 1) / TILE_R, DH / TILE_C);                  // 391 x 4
    dim3 gemmGrdC((NN + TILE_R - 1) / TILE_R, (NC + TILE_C - 1) / TILE_C);   // 391 x 2

    // ---- graph normalization ----
    zero_f32<<<(NN + B - 1) / B, B, 0, stream>>>(deg, NN);
    deg_kernel<<<(EE + B - 1) / B, B, 0, stream>>>(tgt, deg, EE);
    dinv_kernel<<<(NN + B - 1) / B, B, 0, stream>>>(deg, dinv, NN);
    norm_kernel<<<(EE + B - 1) / B, B, 0, stream>>>(src, tgt, dinv, nrm, EE);

    // ---- layer 0: x[50000,1024] -> h0[50000,128] ----
    gemm_bf16_wmma<<<gemmGrdH, gemmBlk, 0, stream>>>(x, W0, b0, h0, NN, DIN, DH);
    encoder_kernel<<<(NN * 4 + B - 1) / B, B, 0, stream>>>(h0, labels, eW0, eb0, enc, NN, DH);
    zero_f32<<<(NN * DH + B - 1) / B, B, 0, stream>>>(h1, NN * DH);
    edge_decoder_scatter<<<EE, 128, 0, stream>>>(enc, nrm, src, tgt, labels, dW0, db0, h1, DH, EE);
    update_kernel<<<NN, 128, 0, stream>>>(h0, h1, deg, DH);
    zero_f32<<<1, B, 0, stream>>>(stats, 2 * DH);
    bn_stats<<<512, DH, 0, stream>>>(h0, stats, NN, DH);
    bn_apply<<<NN, 128, 0, stream>>>(h0, stats, g0, be0, NN, DH);

    // ---- layer 1: h0 -> h1[50000,128] ----
    gemm_bf16_wmma<<<gemmGrdH, gemmBlk, 0, stream>>>(h0, W1, b1, h1, NN, DH, DH);
    encoder_kernel<<<(NN * 4 + B - 1) / B, B, 0, stream>>>(h1, labels, eW1, eb1, enc, NN, DH);
    zero_f32<<<(NN * DH + B - 1) / B, B, 0, stream>>>(h0, NN * DH);
    edge_decoder_scatter<<<EE, 128, 0, stream>>>(enc, nrm, src, tgt, labels, dW1, db1, h0, DH, EE);
    update_kernel<<<NN, 128, 0, stream>>>(h1, h0, deg, DH);
    zero_f32<<<1, B, 0, stream>>>(stats, 2 * DH);
    bn_stats<<<512, DH, 0, stream>>>(h1, stats, NN, DH);
    bn_apply<<<NN, 128, 0, stream>>>(h1, stats, g1, be1, NN, DH);

    // ---- layer 2: h1 -> out[50000,40] ----
    gemm_bf16_wmma<<<gemmGrdC, gemmBlk, 0, stream>>>(h1, W2, b2, h0, NN, DH, NC);
    encoder_kernel<<<(NN * 4 + B - 1) / B, B, 0, stream>>>(h0, labels, eW2, eb2, enc, NN, NC);
    zero_f32<<<(NN * NC + B - 1) / B, B, 0, stream>>>(h1, NN * NC);
    edge_decoder_scatter<<<EE, 64, 0, stream>>>(enc, nrm, src, tgt, labels, dW2, db2, h1, NC, EE);
    final_kernel<<<NN, 64, 0, stream>>>(h0, h1, deg, out, NC);
}
